// Attention_61177514165028
// MI455X (gfx1250) — matrix operand order repaired and checked by reading, not compiled, not run on hardware
//
#include <hip/hip_runtime.h>

#define DEV static __device__ __forceinline__

typedef __attribute__((ext_vector_type(16))) __bf16 v16bf;
typedef __attribute__((ext_vector_type(8)))  float  v8f;

constexpr int BATCH = 2;
constexpr int SEQ   = 2048;
constexpr int DIM   = 1024;
constexpr int HEADS = 16;
constexpr int DHEAD = 64;
constexpr int NQKV  = 3 * DIM;   // 3072
constexpr float SCALE = 0.125f;  // 64^-0.5

// ---------- helpers ----------

DEV unsigned short f2bf(float f) {
  unsigned u = __float_as_uint(f);
  u += 0x7FFFu + ((u >> 16) & 1u);   // round-to-nearest-even
  return (unsigned short)(u >> 16);
}

union FragU { uint4 q[2]; v16bf v; };

// A fragment: 16(M) x 32(K), src row-major bf16, K contiguous.
// lane L: M = L&15 ; elems 0..7 -> K = (L>>4)*8 + e ; elems 8..15 -> K = 16 + (L>>4)*8 + e
DEV v16bf load_frag_a(const unsigned short* p, int ld) {
  int lane = threadIdx.x & 31;
  int row = lane & 15, hf = lane >> 4;
  FragU u;
  u.q[0] = *(const uint4*)(p + row * ld + hf * 8);
  u.q[1] = *(const uint4*)(p + row * ld + hf * 8 + 16);
  return u.v;
}

// B fragment: 32(K) x 16(N), src stored N-major (row n holds contiguous K).
// lane L: N = L&15 ; elems 0..7 -> K = (L>>4)*8 + e ; elems 8..15 -> K = 16 + (L>>4)*8 + e
DEV v16bf load_frag_b(const unsigned short* p, int ld) {
  int lane = threadIdx.x & 31;
  int col = lane & 15, hf = lane >> 4;
  FragU u;
  u.q[0] = *(const uint4*)(p + col * ld + hf * 8);
  u.q[1] = *(const uint4*)(p + col * ld + hf * 8 + 16);
  return u.v;
}

DEV v8f wmma_bf16(v16bf a, v16bf b, v8f c) {
  return __builtin_amdgcn_wmma_f32_16x16x32_bf16(false, a, false, b, (short)0, c,
                                                 false, false);
}

// ---------- kernel 0: weight convert + transpose (LDS-tiled, coalesced) ----------
// in: [R][C] f32 (row-major)  ->  outT: [C][R] bf16
__global__ void __launch_bounds__(256) transpose_bf16_kernel(
    const float* __restrict__ in, unsigned short* __restrict__ outT, int R, int C) {
  __shared__ float tile[32][33];
  int cx = blockIdx.x * 32, ry = blockIdx.y * 32;
  int tx = threadIdx.x & 31, ty = threadIdx.x >> 5;  // 32 x 8
#pragma unroll
  for (int j = 0; j < 32; j += 8)
    tile[ty + j][tx] = in[(size_t)(ry + ty + j) * C + (cx + tx)];
  __syncthreads();
#pragma unroll
  for (int j = 0; j < 32; j += 8)
    outT[(size_t)(cx + ty + j) * R + (ry + tx)] = f2bf(tile[tx][ty + j]);
}

// ---------- kernel 1: LayerNorm -> bf16 ----------

__global__ void __launch_bounds__(256) ln_kernel(const float* __restrict__ x,
                                                 const float* __restrict__ g,
                                                 const float* __restrict__ bta,
                                                 unsigned short* __restrict__ xn) {
  int row = blockIdx.x, tid = threadIdx.x;
  const float4* xr = (const float4*)(x + (size_t)row * DIM);
  float4 v = xr[tid];
  __shared__ float red[256];
  red[tid] = v.x + v.y + v.z + v.w;
  __syncthreads();
  for (int off = 128; off > 0; off >>= 1) {
    if (tid < off) red[tid] += red[tid + off];
    __syncthreads();
  }
  float mu = red[0] * (1.0f / DIM);
  __syncthreads();
  float dx = v.x - mu, dy = v.y - mu, dz = v.z - mu, dw = v.w - mu;
  red[tid] = dx * dx + dy * dy + dz * dz + dw * dw;
  __syncthreads();
  for (int off = 128; off > 0; off >>= 1) {
    if (tid < off) red[tid] += red[tid + off];
    __syncthreads();
  }
  float rstd = rsqrtf(red[0] * (1.0f / DIM) + 1e-5f);
  int idx = tid * 4;
  unsigned short* o = xn + (size_t)row * DIM + idx;
  o[0] = f2bf(dx * rstd * g[idx + 0] + bta[idx + 0]);
  o[1] = f2bf(dy * rstd * g[idx + 1] + bta[idx + 1]);
  o[2] = f2bf(dz * rstd * g[idx + 2] + bta[idx + 2]);
  o[3] = f2bf(dw * rstd * g[idx + 3] + bta[idx + 3]);
}

// ---------- kernel 2: QKV GEMM (32x64/wave, ping-pong pipelined) + scatter ----------

__global__ void __launch_bounds__(128) qkv_gemm_kernel(
    const unsigned short* __restrict__ xn, const unsigned short* __restrict__ wt,
    unsigned short* __restrict__ Qb, unsigned short* __restrict__ Kb,
    unsigned short* __restrict__ Vtb) {
  int w = threadIdx.x >> 5, lane = threadIdx.x & 31;
  int hf = lane >> 4, nl = lane & 15;
  int m0 = blockIdx.y * 128 + w * 32;
  int n0 = blockIdx.x * 64;
  v8f z = {0.f, 0.f, 0.f, 0.f, 0.f, 0.f, 0.f, 0.f};
  v8f acc[2][4] = {{z, z, z, z}, {z, z, z, z}};

  const unsigned short* A0 = xn + (size_t)m0 * DIM;
  const unsigned short* A1 = xn + (size_t)(m0 + 16) * DIM;
  const unsigned short* B  = wt + (size_t)n0 * DIM;

  v16bf a0[2], b0[4], a1[2], b1[4];
  a0[0] = load_frag_a(A0, DIM);
  a0[1] = load_frag_a(A1, DIM);
#pragma unroll
  for (int t = 0; t < 4; ++t) b0[t] = load_frag_b(B + (size_t)(16 * t) * DIM, DIM);

  for (int k0 = 0; k0 < DIM; k0 += 64) {
    int k1 = k0 + 32;
    int k2 = (k0 + 64 < DIM) ? (k0 + 64) : 0;  // last iter: wraps (redundant)
    __builtin_prefetch(B + k1 + 256, 0, 1);
    // load set1 @ k1 while computing with set0 @ k0
    a1[0] = load_frag_a(A0 + k1, DIM);
    a1[1] = load_frag_a(A1 + k1, DIM);
#pragma unroll
    for (int t = 0; t < 4; ++t)
      b1[t] = load_frag_b(B + (size_t)(16 * t) * DIM + k1, DIM);
#pragma unroll
    for (int mi = 0; mi < 2; ++mi)
#pragma unroll
      for (int t = 0; t < 4; ++t)
        acc[mi][t] = wmma_bf16(a0[mi], b0[t], acc[mi][t]);
    // load set0 @ k2 while computing with set1 @ k1
    a0[0] = load_frag_a(A0 + k2, DIM);
    a0[1] = load_frag_a(A1 + k2, DIM);
#pragma unroll
    for (int t = 0; t < 4; ++t)
      b0[t] = load_frag_b(B + (size_t)(16 * t) * DIM + k2, DIM);
#pragma unroll
    for (int mi = 0; mi < 2; ++mi)
#pragma unroll
      for (int t = 0; t < 4; ++t)
        acc[mi][t] = wmma_bf16(a1[mi], b1[t], acc[mi][t]);
  }

#pragma unroll
  for (int mi = 0; mi < 2; ++mi) {
#pragma unroll
    for (int t = 0; t < 4; ++t) {
      int n = n0 + 16 * t + nl;
#pragma unroll
      for (int r = 0; r < 8; ++r) {
        int m = m0 + 16 * mi + r + 8 * hf;
        int bb = m >> 11, i = m & (SEQ - 1);
        float val = acc[mi][t][r];
        if (n < DIM) {
          int h = n >> 6, d = n & 63;
          Qb[(((size_t)(bb * HEADS + h)) * SEQ + i) * DHEAD + d] = f2bf(val * SCALE);
        } else if (n < 2 * DIM) {
          int nn = n - DIM;
          int h = nn >> 6, d = nn & 63;
          Kb[(((size_t)(bb * HEADS + h)) * SEQ + i) * DHEAD + d] = f2bf(val);
        } else {
          int nn = n - 2 * DIM;
          int h = nn >> 6, d = nn & 63;
          Vtb[(((size_t)(bb * HEADS + h)) * DHEAD + d) * SEQ + i] = f2bf(val);
        }
      }
    }
  }
}

// ---------- kernel 3: causal flash attention (K double-buffered, V hoisted) ----------

__global__ void __launch_bounds__(128) attn_kernel(
    const unsigned short* __restrict__ Qb, const unsigned short* __restrict__ Kb,
    const unsigned short* __restrict__ Vtb, unsigned short* __restrict__ Ob) {
  int bh = blockIdx.y;
  int w = threadIdx.x >> 5, lane = threadIdx.x & 31;
  int hf = lane >> 4, nl = lane & 15;
  const unsigned short* Q  = Qb  + (size_t)bh * SEQ * DHEAD;
  const unsigned short* K  = Kb  + (size_t)bh * SEQ * DHEAD;
  const unsigned short* Vt = Vtb + (size_t)bh * DHEAD * SEQ;
  int base = blockIdx.x * 64 + w * 16;  // this wave's 16 query rows

  __shared__ unsigned short Pl[4][16][32];  // per-wave P staging (C-layout -> A-layout)

  v16bf aql = load_frag_a(Q + (size_t)base * DHEAD, DHEAD);       // d 0..31
  v16bf aqh = load_frag_a(Q + (size_t)base * DHEAD + 32, DHEAD);  // d 32..63

  v8f z = {0.f, 0.f, 0.f, 0.f, 0.f, 0.f, 0.f, 0.f};
  v8f o[4] = {z, z, z, z};
  float mrow[8], lrow[8];
#pragma unroll
  for (int r = 0; r < 8; ++r) { mrow[r] = -1e30f; lrow[r] = 0.f; }

  // preload K fragments for chunk 0
  v16bf kb[4];
  kb[0] = load_frag_b(K, DHEAD);
  kb[1] = load_frag_b(K + 32, DHEAD);
  kb[2] = load_frag_b(K + (size_t)16 * DHEAD, DHEAD);
  kb[3] = load_frag_b(K + (size_t)16 * DHEAD + 32, DHEAD);

  for (int j0 = 0; j0 <= base + 15; j0 += 32) {  // N%32==0 -> never OOB
    v8f s0 = z, s1 = z;
    s0 = wmma_bf16(aql, kb[0], s0);
    s0 = wmma_bf16(aqh, kb[1], s0);
    s1 = wmma_bf16(aql, kb[2], s1);
    s1 = wmma_bf16(aqh, kb[3], s1);

    // issue V loads for this chunk early (latency hides behind softmax VALU work)
    v16bf bv[4];
#pragma unroll
    for (int t = 0; t < 4; ++t)
      bv[t] = load_frag_b(Vt + (size_t)(16 * t) * SEQ + j0, SEQ);

    // prefetch next chunk's K fragments (redundant reload on last iter)
    int jn = (j0 + 32 <= base + 15) ? (j0 + 32) : j0;
    kb[0] = load_frag_b(K + (size_t)jn * DHEAD, DHEAD);
    kb[1] = load_frag_b(K + (size_t)jn * DHEAD + 32, DHEAD);
    kb[2] = load_frag_b(K + (size_t)(jn + 16) * DHEAD, DHEAD);
    kb[3] = load_frag_b(K + (size_t)(jn + 16) * DHEAD + 32, DHEAD);

    int col0 = j0 + nl, col1 = j0 + 16 + nl;
#pragma unroll
    for (int r = 0; r < 8; ++r) {
      int row = base + r + 8 * hf;
      float v0 = (col0 > row) ? -1e30f : s0[r];
      float v1 = (col1 > row) ? -1e30f : s1[r];
      float mx = fmaxf(v0, v1);
#pragma unroll
      for (int off = 1; off < 16; off <<= 1) mx = fmaxf(mx, __shfl_xor(mx, off, 32));
      float mnew = fmaxf(mrow[r], mx);
      float al = __expf(mrow[r] - mnew);
      float p0 = __expf(v0 - mnew);
      float p1 = __expf(v1 - mnew);
      float sum = p0 + p1;
#pragma unroll
      for (int off = 1; off < 16; off <<= 1) sum += __shfl_xor(sum, off, 32);
      lrow[r] = lrow[r] * al + sum;
      mrow[r] = mnew;
#pragma unroll
      for (int t = 0; t < 4; ++t) o[t][r] *= al;
      Pl[w][r + 8 * hf][nl]      = f2bf(p0);
      Pl[w][r + 8 * hf][nl + 16] = f2bf(p1);
    }
    // same-wave LDS write -> cross-lane read ordering
    asm volatile("s_wait_dscnt 0" ::: "memory");
    v16bf ap;
    {
      FragU u;
      const unsigned short* pp = &Pl[w][0][0];
      u.q[0] = *(const uint4*)(pp + nl * 32 + hf * 8);
      u.q[1] = *(const uint4*)(pp + nl * 32 + hf * 8 + 16);
      ap = u.v;
    }
#pragma unroll
    for (int t = 0; t < 4; ++t) o[t] = wmma_bf16(ap, bv[t], o[t]);
    asm volatile("s_wait_dscnt 0" ::: "memory");  // WAR guard vs next chunk's P stores
  }

  int b = bh >> 4, h = bh & 15;
#pragma unroll
  for (int t = 0; t < 4; ++t) {
    int d = 16 * t + nl;
#pragma unroll
    for (int r = 0; r < 8; ++r) {
      int row = base + r + 8 * hf;
      float val = o[t][r] / lrow[r];
      Ob[((size_t)(b * SEQ + row)) * DIM + h * DHEAD + d] = f2bf(val);
    }
  }
}

// ---------- kernel 4: output projection GEMM (32x64/wave, ping-pong pipelined) ----------

__global__ void __launch_bounds__(128) out_gemm_kernel(
    const unsigned short* __restrict__ ao, const unsigned short* __restrict__ wt,
    float* __restrict__ out) {
  int w = threadIdx.x >> 5, lane = threadIdx.x & 31;
  int hf = lane >> 4, nl = lane & 15;
  int m0 = blockIdx.y * 128 + w * 32;
  int n0 = blockIdx.x * 64;
  v8f z = {0.f, 0.f, 0.f, 0.f, 0.f, 0.f, 0.f, 0.f};
  v8f acc[2][4] = {{z, z, z, z}, {z, z, z, z}};

  const unsigned short* A0 = ao + (size_t)m0 * DIM;
  const unsigned short* A1 = ao + (size_t)(m0 + 16) * DIM;
  const unsigned short* B  = wt + (size_t)n0 * DIM;

  v16bf a0[2], b0[4], a1[2], b1[4];
  a0[0] = load_frag_a(A0, DIM);
  a0[1] = load_frag_a(A1, DIM);
#pragma unroll
  for (int t = 0; t < 4; ++t) b0[t] = load_frag_b(B + (size_t)(16 * t) * DIM, DIM);

  for (int k0 = 0; k0 < DIM; k0 += 64) {
    int k1 = k0 + 32;
    int k2 = (k0 + 64 < DIM) ? (k0 + 64) : 0;
    __builtin_prefetch(B + k1 + 256, 0, 1);
    a1[0] = load_frag_a(A0 + k1, DIM);
    a1[1] = load_frag_a(A1 + k1, DIM);
#pragma unroll
    for (int t = 0; t < 4; ++t)
      b1[t] = load_frag_b(B + (size_t)(16 * t) * DIM + k1, DIM);
#pragma unroll
    for (int mi = 0; mi < 2; ++mi)
#pragma unroll
      for (int t = 0; t < 4; ++t)
        acc[mi][t] = wmma_bf16(a0[mi], b0[t], acc[mi][t]);
    a0[0] = load_frag_a(A0 + k2, DIM);
    a0[1] = load_frag_a(A1 + k2, DIM);
#pragma unroll
    for (int t = 0; t < 4; ++t)
      b0[t] = load_frag_b(B + (size_t)(16 * t) * DIM + k2, DIM);
#pragma unroll
    for (int mi = 0; mi < 2; ++mi)
#pragma unroll
      for (int t = 0; t < 4; ++t)
        acc[mi][t] = wmma_bf16(a1[mi], b1[t], acc[mi][t]);
  }

#pragma unroll
  for (int mi = 0; mi < 2; ++mi)
#pragma unroll
    for (int t = 0; t < 4; ++t) {
      int n = n0 + 16 * t + nl;
#pragma unroll
      for (int r = 0; r < 8; ++r) {
        int m = m0 + 16 * mi + r + 8 * hf;
        out[(size_t)m * DIM + n] = acc[mi][t][r];
      }
    }
}

// ---------- host launcher ----------

extern "C" void kernel_launch(void* const* d_in, const int* in_sizes, int n_in,
                              void* d_out, int out_size, void* d_ws, size_t ws_size,
                              hipStream_t stream) {
  const float* x     = (const float*)d_in[0];
  const float* ln_w  = (const float*)d_in[1];
  const float* ln_b  = (const float*)d_in[2];
  const float* w_qkv = (const float*)d_in[3];
  const float* w_out = (const float*)d_in[4];
  float* out = (float*)d_out;

  char* ws = (char*)d_ws;
  unsigned short* xn    = (unsigned short*)ws; ws += (size_t)BATCH * SEQ * DIM * 2;
  unsigned short* wqkvT = (unsigned short*)ws; ws += (size_t)NQKV * DIM * 2;
  unsigned short* woutT = (unsigned short*)ws; ws += (size_t)DIM * DIM * 2;
  unsigned short* Qb    = (unsigned short*)ws; ws += (size_t)BATCH * HEADS * SEQ * DHEAD * 2;
  unsigned short* Kb    = (unsigned short*)ws; ws += (size_t)BATCH * HEADS * SEQ * DHEAD * 2;
  unsigned short* Vtb   = (unsigned short*)ws; ws += (size_t)BATCH * HEADS * DHEAD * SEQ * 2;
  unsigned short* Ob    = (unsigned short*)ws; ws += (size_t)BATCH * SEQ * DIM * 2;

  transpose_bf16_kernel<<<dim3(NQKV / 32, DIM / 32), 256, 0, stream>>>(w_qkv, wqkvT,
                                                                       DIM, NQKV);
  transpose_bf16_kernel<<<dim3(DIM / 32, DIM / 32), 256, 0, stream>>>(w_out, woutT,
                                                                      DIM, DIM);
  ln_kernel<<<BATCH * SEQ, 256, 0, stream>>>(x, ln_w, ln_b, xn);
  qkv_gemm_kernel<<<dim3(NQKV / 64, (BATCH * SEQ) / 128), 128, 0, stream>>>(
      xn, wqkvT, Qb, Kb, Vtb);
  attn_kernel<<<dim3(SEQ / 64, BATCH * HEADS), 128, 0, stream>>>(Qb, Kb, Vtb, Ob);
  out_gemm_kernel<<<dim3(DIM / 64, (BATCH * SEQ) / 128), 128, 0, stream>>>(
      Ob, woutT, out);
}
